// GMS_32401233281697
// MI455X (gfx1250) — compile-verified, matrix operand order checked by reading, not run on hardware
//
#include <hip/hip_runtime.h>
#include <hip/hip_bf16.h>

#define DIM 128
#define NV 50000
#define NC 210000
#define NE 300000
#define NROUNDS 16

typedef __attribute__((ext_vector_type(16))) _Float16 v16h;
typedef __attribute__((ext_vector_type(8)))  float    v8f;
typedef __attribute__((ext_vector_type(4)))  unsigned int v4u;
typedef __attribute__((ext_vector_type(8)))  int      v8i;
typedef __attribute__((ext_vector_type(4)))  int      v4i;

#if __has_builtin(__builtin_amdgcn_tensor_load_to_lds)
#define HAVE_TDM 1
#warning "CDNA5 TDM: __builtin_amdgcn_tensor_load_to_lds AVAILABLE - TDM path compiled"
#else
#define HAVE_TDM 0
#warning "CDNA5 TDM: builtin NOT available - cooperative-copy fallback compiled"
#endif

__device__ __forceinline__ v8f wmma_f32_16x16x32_f16(v16h a, v16h b, v8f c) {
  // (neg_a, A, neg_b, B, c_mod, C, reuse_a, reuse_b)
  return __builtin_amdgcn_wmma_f32_16x16x32_f16(false, a, false, b, (short)0, c, false, false);
}

__device__ __forceinline__ float sigmoidf_(float x) {
  return 1.0f / (1.0f + __expf(-x));
}

// ---------------------------------------------------------------------------
// TDM 2-D tile load: Global -> LDS, data_size = sizeof(T).
// Call from ONE wave only (TDM ignores EXEC; wave-level op, TENSORcnt).
// Descriptor per CDNA5 ISA ch.8: group0 {count, lds_addr, global_addr, type=2},
// group1 {data_size, tensor_dim0/1, tile_dim0/1, tensor_dim0_stride}.
// ---------------------------------------------------------------------------
template <typename T>
__device__ __forceinline__ void tdm_load_2d(T* ldsDst, const T* gsrc,
                                            unsigned widthElems, unsigned heightRows,
                                            unsigned strideElems) {
#if HAVE_TDM
  constexpr unsigned dsLog2 = (sizeof(T) == 1) ? 0u : (sizeof(T) == 2) ? 1u : 2u;
  const unsigned ldsOff = (unsigned)(unsigned long long)ldsDst;  // addr[31:0] = LDS offset
  const unsigned long long ga = (unsigned long long)gsrc;
  v4u g0;
  g0[0] = 1u;                                   // count=1, user mode, no gather
  g0[1] = ldsOff;                               // lds_addr
  g0[2] = (unsigned)ga;                         // global_addr[31:0]
  g0[3] = (unsigned)((ga >> 32) & 0x1FFFFFFu)   // global_addr[56:32]
          | (2u << 30);                         // type = 2 ("image")
  v8i g1;
  g1[0] = (int)(dsLog2 << 16);                  // workgroup_mask=0, data_size
  g1[1] = (int)((widthElems & 0xFFFFu) << 16);  // tensor_dim0[15:0]
  g1[2] = (int)(((widthElems >> 16) & 0xFFFFu) |
                ((heightRows & 0xFFFFu) << 16));        // dim0 hi | tensor_dim1 lo
  g1[3] = (int)(((heightRows >> 16) & 0xFFFFu) |
                ((widthElems & 0xFFFFu) << 16));        // dim1 hi | tile_dim0
  g1[4] = (int)(heightRows & 0xFFFFu);                  // tile_dim1 (tile_dim2 = 0)
  g1[5] = (int)strideElems;                             // tensor_dim0_stride[31:0]
  g1[6] = 0;
  g1[7] = 0;
  v4i z4 = {0, 0, 0, 0};
#if defined(__clang_major__) && (__clang_major__ >= 23)
  v8i z8 = {0, 0, 0, 0, 0, 0, 0, 0};
  __builtin_amdgcn_tensor_load_to_lds(g0, g1, z4, z4, z8, 0);
#else
  __builtin_amdgcn_tensor_load_to_lds(g0, g1, z4, z4, 0);
#endif
#else
  // Fallback: cooperative copy by this wave.
  const int lane = threadIdx.x & 31;
  for (unsigned r = 0; r < heightRows; ++r)
    for (unsigned i = lane; i < widthElems; i += 32)
      ldsDst[r * widthElems + i] = gsrc[(size_t)r * strideElems + i];
#endif
}

__device__ __forceinline__ void tdm_wait(int n) {
#if HAVE_TDM
  if (n == 0)
    __builtin_amdgcn_s_wait_tensorcnt(0);
  else
    __builtin_amdgcn_s_wait_tensorcnt(1);
#else
  (void)n;
#endif
}

// One 16x16 output tile of one dense layer step: acc += A(16x32) * B(32x16)
// over K=128, A/B both LDS-resident f16.
__device__ __forceinline__ v8f mlp_tile_k128(const _Float16* __restrict__ aTile,
                                             const _Float16* __restrict__ bRow,
                                             int arow, int ak0) {
  v8f acc = {};
#pragma unroll
  for (int kk = 0; kk < 4; ++kk) {
    v16h a = *(const v16h*)(aTile + arow * DIM + kk * 32 + ak0);
    v16h b = *(const v16h*)(bRow + kk * 32);
    acc = wmma_f32_16x16x32_f16(a, b, acc);
  }
  return acc;
}

// ---------------------------------------------------------------------------
// f32 -> f16 weight conversion
// ---------------------------------------------------------------------------
__global__ void cvt_f16_kernel(const float* __restrict__ src, _Float16* __restrict__ dst, int n) {
  int i = blockIdx.x * blockDim.x + threadIdx.x;
  int stride = gridDim.x * blockDim.x;
  for (; i < n; i += stride) dst[i] = (_Float16)src[i];
}

// ---------------------------------------------------------------------------
// zero-fill
// ---------------------------------------------------------------------------
__global__ void fill0_kernel(float* __restrict__ p, size_t n) {
  size_t i = (size_t)blockIdx.x * blockDim.x + threadIdx.x;
  size_t stride = (size_t)gridDim.x * blockDim.x;
  for (; i < n; i += stride) p[i] = 0.0f;
}

// ---------------------------------------------------------------------------
// state init: h = initW + initb (tiled), hH = f16 shadow, c = 0
// ---------------------------------------------------------------------------
__global__ void init_state_kernel(float* __restrict__ h, _Float16* __restrict__ hH,
                                  float* __restrict__ c,
                                  const float* __restrict__ initW,
                                  const float* __restrict__ initb, int nrows) {
  size_t i = (size_t)blockIdx.x * blockDim.x + threadIdx.x;
  size_t stride = (size_t)gridDim.x * blockDim.x;
  size_t n = (size_t)nrows * DIM;
  for (; i < n; i += stride) {
    int d = (int)(i & (DIM - 1));
    float v = initW[d] + initb[d];
    h[i] = v;
    hH[i] = (_Float16)v;
    c[i] = 0.0f;
  }
}

// ---------------------------------------------------------------------------
// Fused 3-layer MLP (Linear-ReLU-Linear-ReLU-Linear), DIM=128, f16 input.
// grid.y in {0,1} selects MLP index (mlpBase + y) and output buffer.
// Weights (3x128x128 f16) TDM-loaded to LDS once; activation 16x128 tiles
// TDM-loaded with double buffering (s_wait_tensorcnt 1 overlap). Buffer
// selection via ternary of LDS pointers so ds_load_b128 is preserved.
// ---------------------------------------------------------------------------
__global__ void __launch_bounds__(256)
mlp_msg_kernel(const _Float16* __restrict__ srcH, float* __restrict__ dstP,
               float* __restrict__ dstN, const _Float16* __restrict__ msgW,
               const float* __restrict__ mb1, const float* __restrict__ mb2,
               const float* __restrict__ mb3, int mlpBase, int nrows) {
  extern __shared__ char smem[];
  _Float16* sW   = (_Float16*)smem;           // 3*128*128 f16 = 96 KB
  _Float16* sIn0 = sW + 3 * DIM * DIM;        // 16*128 f16
  _Float16* sIn1 = sIn0 + 16 * DIM;           // 16*128 f16
  _Float16* sT0  = sIn1 + 16 * DIM;           // 16*128 f16
  _Float16* sT1  = sT0 + 16 * DIM;            // 16*128 f16
  float*    sB   = (float*)(sT1 + 16 * DIM);  // 3*128 f32

  const int m = mlpBase + (int)blockIdx.y;
  float* dst = blockIdx.y ? dstN : dstP;
  const int tid = threadIdx.x;
  const int wave = tid >> 5, lane = tid & 31;

  // Stage all 3 layer weight matrices via one TDM 2-D transfer (384x128 f16).
  if (wave == 0)
    tdm_load_2d(sW, msgW + (size_t)m * 3 * DIM * DIM, DIM, 3 * DIM, DIM);
  for (int i = tid; i < DIM; i += blockDim.x) {
    sB[i]           = mb1[m * DIM + i];
    sB[DIM + i]     = mb2[m * DIM + i];
    sB[2 * DIM + i] = mb3[m * DIM + i];
  }

  const int colBase = wave * 16;
  const int arow = lane & 15;
  const int ak0  = (lane >> 4) * 16;
  const int n    = colBase + (lane & 15);
  const int rbase = (lane < 16) ? 0 : 8;
  const _Float16* bRow0 = sW + (size_t)(colBase + (lane & 15)) * DIM + ak0;

  const int nblk = nrows >> 4;
  int buf = 0;

  // Prologue: issue first activation tile.
  if ((int)blockIdx.x < nblk && wave == 0)
    tdm_load_2d(sIn0, srcH + (size_t)blockIdx.x * 16 * DIM, DIM, 16, DIM);

  for (int rb = blockIdx.x; rb < nblk; rb += gridDim.x) {
    const int rbn = rb + (int)gridDim.x;
    if (wave == 0) {
      if (rbn < nblk) {
        tdm_load_2d(buf ? sIn0 : sIn1, srcH + (size_t)rbn * 16 * DIM, DIM, 16, DIM);
        tdm_wait(1);   // current tile (and weights) done; next may be in flight
      } else {
        tdm_wait(0);
      }
    }
    __syncthreads();

    const _Float16* in = buf ? sIn1 : sIn0;  // LDS-pointer select (keeps ds_load)

    // ---- layer 0: in -> sT0 (ReLU) ----
    {
      v8f acc = mlp_tile_k128(in, bRow0, arow, ak0);
      const float bias = sB[n];
#pragma unroll
      for (int r = 0; r < 8; ++r) {
        float v = acc[r] + bias;
        v = v > 0.0f ? v : 0.0f;
        sT0[(rbase + r) * DIM + n] = (_Float16)v;
      }
    }
    __syncthreads();
    // ---- layer 1: sT0 -> sT1 (ReLU) ----
    {
      v8f acc = mlp_tile_k128(sT0, bRow0 + DIM * DIM, arow, ak0);
      const float bias = sB[DIM + n];
#pragma unroll
      for (int r = 0; r < 8; ++r) {
        float v = acc[r] + bias;
        v = v > 0.0f ? v : 0.0f;
        sT1[(rbase + r) * DIM + n] = (_Float16)v;
      }
    }
    __syncthreads();
    // ---- layer 2: sT1 -> global (no activation) ----
    {
      v8f acc = mlp_tile_k128(sT1, bRow0 + 2 * DIM * DIM, arow, ak0);
      const float bias = sB[2 * DIM + n];
      float* drow = dst + (size_t)rb * 16 * DIM;
#pragma unroll
      for (int r = 0; r < 8; ++r)
        drow[(size_t)(rbase + r) * DIM + n] = acc[r] + bias;
    }
    __syncthreads();
    buf ^= 1;
  }
}

// ---------------------------------------------------------------------------
// Edge scatter-add: acc[dstIdx[e]] += msg[srcIdx[e]]  (one wave per edge,
// 4 floats per lane, L2-resident f32 atomics; prefetch next edge's row).
// ---------------------------------------------------------------------------
__global__ void __launch_bounds__(256)
scatter_add_kernel(const float* __restrict__ msg, const int* __restrict__ srcIdx,
                   const int* __restrict__ dstIdx, float* __restrict__ acc,
                   int nEdges) {
  const int wave = (int)((blockIdx.x * blockDim.x + threadIdx.x) >> 5);
  const int lane = threadIdx.x & 31;
  const int nW = (int)((gridDim.x * blockDim.x) >> 5);
  for (int e = wave; e < nEdges; e += nW) {
    const int s = srcIdx[e];
    const int d = dstIdx[e];
    if (e + nW < nEdges) {
      int sn = srcIdx[e + nW];
      __builtin_prefetch(msg + (size_t)sn * DIM + lane * 4, 0, 1);
    }
    const float4 v = *(const float4*)(msg + (size_t)s * DIM + lane * 4);
    float* out = acc + (size_t)d * DIM + lane * 4;
    atomicAdd(out + 0, v.x);
    atomicAdd(out + 1, v.y);
    atomicAdd(out + 2, v.z);
    atomicAdd(out + 3, v.w);
  }
}

// ---------------------------------------------------------------------------
// Fused LSTM cell: gates = x@Wih.T + h@Whh.T + bih + bhh (WMMA; both 512x128
// weight matrices TDM-staged in LDS = 256 KB), then elementwise h,c update.
// x (f32 segment-sum) TDM-loaded as 4-byte tile then converted; h read from
// its f16 shadow. Writes h (f32), hH (f16 shadow), c (f32) in place.
// ---------------------------------------------------------------------------
__global__ void __launch_bounds__(256)
lstm_kernel(const float* __restrict__ x, float* __restrict__ h,
            _Float16* __restrict__ hH, float* __restrict__ c,
            const _Float16* __restrict__ Wih, const _Float16* __restrict__ Whh,
            const float* __restrict__ bih, const float* __restrict__ bhh, int nrows) {
  extern __shared__ char smem[];
  _Float16* sWi = (_Float16*)smem;            // 512*128 f16 = 128 KB
  _Float16* sWh = sWi + 512 * DIM;            // 512*128 f16 = 128 KB
  _Float16* sX  = sWh + 512 * DIM;            // 16*128 f16
  _Float16* sH  = sX + 16 * DIM;              // 16*128 f16
  float* sXF    = (float*)(sH + 16 * DIM);    // 16*128 f32 (raw x tile)
  float* sG     = sXF + 16 * DIM;             // 16*512 f32 = 32 KB
  float* sBias  = sG + 16 * 512;              // 512 f32

  const int tid = threadIdx.x, wave = tid >> 5, lane = tid & 31;
  if (wave == 0) {
    tdm_load_2d(sWi, Wih, DIM, 512, DIM);
    tdm_load_2d(sWh, Whh, DIM, 512, DIM);
  }
  for (int i = tid; i < 512; i += blockDim.x) sBias[i] = bih[i] + bhh[i];

  const int arow = lane & 15;
  const int ak0  = (lane >> 4) * 16;
  const int nblk = nrows >> 4;
  for (int rb = blockIdx.x; rb < nblk; rb += gridDim.x) {
    const size_t base = (size_t)rb * 16 * DIM;
    if (wave == 0) {
      tdm_load_2d(sXF, x + base, DIM, 16, DIM);
      tdm_load_2d(sH, hH + base, DIM, 16, DIM);
      tdm_wait(0);
    }
    __syncthreads();
    for (int i = tid; i < 16 * DIM; i += blockDim.x) sX[i] = (_Float16)sXF[i];
    __syncthreads();

#pragma unroll
    for (int t = 0; t < 4; ++t) {
      const int ct = wave * 4 + t;  // col tile 0..31 of 512-wide gates
      const _Float16* Bi = sWi + (size_t)(ct * 16 + (lane & 15)) * DIM + ak0;
      const _Float16* Bh = sWh + (size_t)(ct * 16 + (lane & 15)) * DIM + ak0;
      v8f acc = {};
#pragma unroll
      for (int kk = 0; kk < 4; ++kk) {
        v16h ax = *(const v16h*)(sX + arow * DIM + kk * 32 + ak0);
        v16h bi = *(const v16h*)(Bi + kk * 32);
        acc = wmma_f32_16x16x32_f16(ax, bi, acc);
        v16h ah = *(const v16h*)(sH + arow * DIM + kk * 32 + ak0);
        v16h bh = *(const v16h*)(Bh + kk * 32);
        acc = wmma_f32_16x16x32_f16(ah, bh, acc);
      }
      const int n = ct * 16 + (lane & 15);
      const int rbase = (lane < 16) ? 0 : 8;
      const float bias = sBias[n];
#pragma unroll
      for (int r = 0; r < 8; ++r) sG[(rbase + r) * 512 + n] = acc[r] + bias;
    }
    __syncthreads();

    // elementwise LSTM update (gate order i, f, g, o)
    for (int i = tid; i < 16 * DIM; i += blockDim.x) {
      const int r = i >> 7, d = i & 127;
      const float gi = sG[r * 512 + d];
      const float gf = sG[r * 512 + 128 + d];
      const float gg = sG[r * 512 + 256 + d];
      const float go = sG[r * 512 + 384 + d];
      const float cold = c[base + i];
      const float cnew = sigmoidf_(gf) * cold + sigmoidf_(gi) * tanhf(gg);
      const float hnew = sigmoidf_(go) * tanhf(cnew);
      c[base + i] = cnew;
      h[base + i] = hnew;
      hH[base + i] = (_Float16)hnew;
    }
    __syncthreads();
  }
}

// ---------------------------------------------------------------------------
// Vote head: Linear-ReLU-Linear-ReLU-Linear(128->1), f16 input.
// ---------------------------------------------------------------------------
__global__ void __launch_bounds__(256)
vote_kernel(const _Float16* __restrict__ srcH, float* __restrict__ out,
            const _Float16* __restrict__ W12, const float* __restrict__ b1,
            const float* __restrict__ b2, const float* __restrict__ W3,
            const float* __restrict__ b3, int nrows) {
  extern __shared__ char smem[];
  _Float16* sW  = (_Float16*)smem;          // 2*128*128 f16
  _Float16* sIn = sW + 2 * DIM * DIM;       // 16*128 f16
  _Float16* sT0 = sIn + 16 * DIM;           // 16*128 f16
  _Float16* sT1 = sT0 + 16 * DIM;           // 16*128 f16
  float* sB  = (float*)(sT1 + 16 * DIM);    // 2*128
  float* sW3 = sB + 2 * DIM;                // 128

  const int tid = threadIdx.x;
  const int wave = tid >> 5, lane = tid & 31;
  if (wave == 0) tdm_load_2d(sW, W12, DIM, 2 * DIM, DIM);
  for (int i = tid; i < DIM; i += blockDim.x) {
    sB[i]       = b1[i];
    sB[DIM + i] = b2[i];
    sW3[i]      = W3[i];
  }

  const int colBase = wave * 16;
  const int arow = lane & 15;
  const int ak0  = (lane >> 4) * 16;
  const int n    = colBase + (lane & 15);
  const int rbase = (lane < 16) ? 0 : 8;
  const _Float16* bRow0 = sW + (size_t)(colBase + (lane & 15)) * DIM + ak0;

  const int nblk = nrows >> 4;
  for (int rb = blockIdx.x; rb < nblk; rb += gridDim.x) {
    if (wave == 0) {
      tdm_load_2d(sIn, srcH + (size_t)rb * 16 * DIM, DIM, 16, DIM);
      tdm_wait(0);
    }
    __syncthreads();

    // layer 0: sIn -> sT0 (ReLU)
    {
      v8f acc = mlp_tile_k128(sIn, bRow0, arow, ak0);
      const float bias = sB[n];
#pragma unroll
      for (int r = 0; r < 8; ++r) {
        float v = acc[r] + bias;
        v = v > 0.0f ? v : 0.0f;
        sT0[(rbase + r) * DIM + n] = (_Float16)v;
      }
    }
    __syncthreads();
    // layer 1: sT0 -> sT1 (ReLU)
    {
      v8f acc = mlp_tile_k128(sT0, bRow0 + DIM * DIM, arow, ak0);
      const float bias = sB[DIM + n];
#pragma unroll
      for (int r = 0; r < 8; ++r) {
        float v = acc[r] + bias;
        v = v > 0.0f ? v : 0.0f;
        sT1[(rbase + r) * DIM + n] = (_Float16)v;
      }
    }
    __syncthreads();
    // final 128 -> 1 dot per row
    if (tid < 16) {
      float s = 0.0f;
      for (int d = 0; d < DIM; ++d) s += (float)sT1[tid * DIM + d] * sW3[d];
      out[rb * 16 + tid] = s + b3[0];
    }
    __syncthreads();
  }
}

// ---------------------------------------------------------------------------
// host-side launch
// ---------------------------------------------------------------------------
extern "C" void kernel_launch(void* const* d_in, const int* in_sizes, int n_in,
                              void* d_out, int out_size, void* d_ws, size_t ws_size,
                              hipStream_t stream) {
  (void)in_sizes; (void)n_in; (void)out_size; (void)ws_size;

  const int*   pos_var    = (const int*)d_in[0];
  const int*   pos_clause = (const int*)d_in[1];
  const int*   neg_var    = (const int*)d_in[2];
  const int*   neg_clause = (const int*)d_in[3];
  const float* L_init_W   = (const float*)d_in[4];
  const float* L_init_b   = (const float*)d_in[5];
  const float* C_init_W   = (const float*)d_in[6];
  const float* C_init_b   = (const float*)d_in[7];
  const float* msg_W1     = (const float*)d_in[8];
  const float* msg_b1     = (const float*)d_in[9];
  const float* msg_W2     = (const float*)d_in[10];
  const float* msg_b2     = (const float*)d_in[11];
  const float* msg_W3     = (const float*)d_in[12];
  const float* msg_b3     = (const float*)d_in[13];
  const float* L_Wih      = (const float*)d_in[14];
  const float* L_Whh      = (const float*)d_in[15];
  const float* L_bih      = (const float*)d_in[16];
  const float* L_bhh      = (const float*)d_in[17];
  const float* C_Wih      = (const float*)d_in[18];
  const float* C_Whh      = (const float*)d_in[19];
  const float* C_bih      = (const float*)d_in[20];
  const float* C_bhh      = (const float*)d_in[21];
  const float* vote_W1    = (const float*)d_in[22];
  const float* vote_b1    = (const float*)d_in[23];
  const float* vote_W2    = (const float*)d_in[24];
  const float* vote_b2    = (const float*)d_in[25];
  const float* vote_W3    = (const float*)d_in[26];
  const float* vote_b3    = (const float*)d_in[27];

  // ---- workspace carving ----
  char* ws = (char*)d_ws;
  size_t off = 0;
  auto carve = [&](size_t bytes) -> void* {
    void* p = ws + off;
    off = (off + bytes + 255) & ~(size_t)255;
    return p;
  };
  _Float16* msgW_h  = (_Float16*)carve((size_t)12 * DIM * DIM * sizeof(_Float16));
  _Float16* L_Wih_h = (_Float16*)carve((size_t)512 * DIM * sizeof(_Float16));
  _Float16* L_Whh_h = (_Float16*)carve((size_t)512 * DIM * sizeof(_Float16));
  _Float16* C_Wih_h = (_Float16*)carve((size_t)512 * DIM * sizeof(_Float16));
  _Float16* C_Whh_h = (_Float16*)carve((size_t)512 * DIM * sizeof(_Float16));
  _Float16* voteW_h = (_Float16*)carve((size_t)2 * DIM * DIM * sizeof(_Float16));
  float* L_h  = (float*)carve((size_t)NV * DIM * sizeof(float));
  float* L_c  = (float*)carve((size_t)NV * DIM * sizeof(float));
  float* C_h  = (float*)carve((size_t)NC * DIM * sizeof(float));
  float* C_c  = (float*)carve((size_t)NC * DIM * sizeof(float));
  _Float16* L_hH = (_Float16*)carve((size_t)NV * DIM * sizeof(_Float16));
  _Float16* C_hH = (_Float16*)carve((size_t)NC * DIM * sizeof(_Float16));
  float* Lp = (float*)carve((size_t)NV * DIM * sizeof(float));
  float* Ln = (float*)carve((size_t)NV * DIM * sizeof(float));
  float* Cp = (float*)carve((size_t)NC * DIM * sizeof(float));
  float* Cn = (float*)carve((size_t)NC * DIM * sizeof(float));
  float* LC = (float*)carve((size_t)NC * DIM * sizeof(float));
  float* CL = (float*)carve((size_t)NV * DIM * sizeof(float));

  // ---- weight conversion to f16 ----
  const int CVT_BLK = 256;
  for (int m = 0; m < 4; ++m) {
    cvt_f16_kernel<<<64, CVT_BLK, 0, stream>>>(msg_W1 + (size_t)m * DIM * DIM,
                                               msgW_h + ((size_t)m * 3 + 0) * DIM * DIM, DIM * DIM);
    cvt_f16_kernel<<<64, CVT_BLK, 0, stream>>>(msg_W2 + (size_t)m * DIM * DIM,
                                               msgW_h + ((size_t)m * 3 + 1) * DIM * DIM, DIM * DIM);
    cvt_f16_kernel<<<64, CVT_BLK, 0, stream>>>(msg_W3 + (size_t)m * DIM * DIM,
                                               msgW_h + ((size_t)m * 3 + 2) * DIM * DIM, DIM * DIM);
  }
  cvt_f16_kernel<<<256, CVT_BLK, 0, stream>>>(L_Wih, L_Wih_h, 512 * DIM);
  cvt_f16_kernel<<<256, CVT_BLK, 0, stream>>>(L_Whh, L_Whh_h, 512 * DIM);
  cvt_f16_kernel<<<256, CVT_BLK, 0, stream>>>(C_Wih, C_Wih_h, 512 * DIM);
  cvt_f16_kernel<<<256, CVT_BLK, 0, stream>>>(C_Whh, C_Whh_h, 512 * DIM);
  cvt_f16_kernel<<<64, CVT_BLK, 0, stream>>>(vote_W1, voteW_h, DIM * DIM);
  cvt_f16_kernel<<<64, CVT_BLK, 0, stream>>>(vote_W2, voteW_h + DIM * DIM, DIM * DIM);

  // ---- state init ----
  init_state_kernel<<<1024, 256, 0, stream>>>(L_h, L_hH, L_c, L_init_W, L_init_b, NV);
  init_state_kernel<<<2048, 256, 0, stream>>>(C_h, C_hH, C_c, C_init_W, C_init_b, NC);

  const size_t MLP_LDS  = (size_t)3 * DIM * DIM * 2 + 4 * 16 * DIM * 2 + 3 * DIM * 4;
  const size_t LSTM_LDS = (size_t)2 * 512 * DIM * 2 + 2 * 16 * DIM * 2 +
                          16 * DIM * 4 + 16 * 512 * 4 + 512 * 4;
  const size_t VOTE_LDS = (size_t)2 * DIM * DIM * 2 + 3 * 16 * DIM * 2 + 2 * DIM * 4 + DIM * 4;

  for (int r = 0; r < NROUNDS; ++r) {
    // literal -> clause
    mlp_msg_kernel<<<dim3(1024, 2), 256, MLP_LDS, stream>>>(
        L_hH, Lp, Ln, msgW_h, msg_b1, msg_b2, msg_b3, 0, NV);
    fill0_kernel<<<2048, 256, 0, stream>>>(LC, (size_t)NC * DIM);
    scatter_add_kernel<<<4096, 256, 0, stream>>>(Lp, pos_var, pos_clause, LC, NE);
    scatter_add_kernel<<<4096, 256, 0, stream>>>(Ln, neg_var, neg_clause, LC, NE);
    lstm_kernel<<<1024, 256, LSTM_LDS, stream>>>(LC, C_h, C_hH, C_c, C_Wih_h, C_Whh_h,
                                                 C_bih, C_bhh, NC);
    // clause -> literal
    mlp_msg_kernel<<<dim3(1024, 2), 256, MLP_LDS, stream>>>(
        C_hH, Cp, Cn, msgW_h, msg_b1, msg_b2, msg_b3, 2, NC);
    fill0_kernel<<<1024, 256, 0, stream>>>(CL, (size_t)NV * DIM);
    scatter_add_kernel<<<4096, 256, 0, stream>>>(Cp, pos_clause, pos_var, CL, NE);
    scatter_add_kernel<<<4096, 256, 0, stream>>>(Cn, neg_clause, neg_var, CL, NE);
    lstm_kernel<<<1024, 256, LSTM_LDS, stream>>>(CL, L_h, L_hH, L_c, L_Wih_h, L_Whh_h,
                                                 L_bih, L_bhh, NV);
  }

  vote_kernel<<<1024, 256, VOTE_LDS, stream>>>(L_hH, (float*)d_out, voteW_h,
                                               vote_b1, vote_b2, vote_W3, vote_b3, NV);
}
